// MoELayer_7687991460157
// MI455X (gfx1250) — matrix operand order repaired and checked by reading, not compiled, not run on hardware
//
#include <hip/hip_runtime.h>
#include <hip/hip_bf16.h>
#include <math.h>

// ---------------------------------------------------------------------------
// MoE layer for MI455X (gfx1250): top-2 expert routing, bf16 WMMA GEMMs.
//   T=1024 B=4 D=1024 E=8 K=2 H=2048 O=1024, N = 4096 tokens.
// Top-2-only FLOPs: 2 * 8192 * 1024 * 2048 * 2 ~= 68.7 GFLOP (matrix-core
// bound). bf16 weights (64MB) are L2-resident across all M-tiles.
// Each wave: 16x32 output tile = 2 accumulators sharing one A fragment
// (halves LDS A traffic, doubles WMMA issue density).
// ---------------------------------------------------------------------------

#define Ntok 4096
#define Ddim 1024
#define Hdim 2048
#define Odim 1024
#define Eexp 8

typedef __bf16 v16bf __attribute__((ext_vector_type(16)));
typedef __bf16 bf4   __attribute__((ext_vector_type(4)));
typedef float  v8f   __attribute__((ext_vector_type(8)));

union ABFrag { v16bf v; uint4 u[2]; };

__device__ __forceinline__ float gelu_exact(float v) {
    return 0.5f * v * (1.0f + erff(v * 0.70710678118654752f));
}

__device__ __forceinline__ float wave_reduce_add(float s) {
#pragma unroll
    for (int m = 16; m > 0; m >>= 1) s += __shfl_xor(s, m, 32);
    return s;
}

// ---------------------------------------------------------------------------
// Kernel 1: LayerNorm -> bf16 xn, router logits, top-2 softmax, expert lists.
// One block (256 threads = 8 waves) per token.
// ---------------------------------------------------------------------------
__global__ __launch_bounds__(256) void ln_router_kernel(
    const float* __restrict__ x, const float* __restrict__ seq_mask,
    const float* __restrict__ ln_scale, const float* __restrict__ ln_bias,
    const float* __restrict__ router_w,
    __bf16* __restrict__ xn, float* __restrict__ gate_out,
    int* __restrict__ counters, int* __restrict__ tok_list,
    int* __restrict__ slot_list, float* __restrict__ prob_list)
{
    const int t   = blockIdx.x;
    const int tid = threadIdx.x;
    __shared__ float sx[Ddim];
    __shared__ float sred[8];
    __shared__ float slog[8];

    float4 v = ((const float4*)(x + (size_t)t * Ddim))[tid];

    // mean
    float s = wave_reduce_add(v.x + v.y + v.z + v.w);
    if ((tid & 31) == 0) sred[tid >> 5] = s;
    __syncthreads();
    float tot = 0.f;
#pragma unroll
    for (int i = 0; i < 8; ++i) tot += sred[i];
    const float mean = tot * (1.0f / Ddim);
    __syncthreads();

    // variance
    float dx = v.x - mean, dy = v.y - mean, dz = v.z - mean, dw = v.w - mean;
    float s2 = wave_reduce_add(dx * dx + dy * dy + dz * dz + dw * dw);
    if ((tid & 31) == 0) sred[tid >> 5] = s2;
    __syncthreads();
    float tot2 = 0.f;
#pragma unroll
    for (int i = 0; i < 8; ++i) tot2 += sred[i];
    const float rs = rsqrtf(tot2 * (1.0f / Ddim) + 1e-5f);

    float4 sc = ((const float4*)ln_scale)[tid];
    float4 bi = ((const float4*)ln_bias)[tid];
    float n0 = dx * rs * sc.x + bi.x;
    float n1 = dy * rs * sc.y + bi.y;
    float n2 = dz * rs * sc.z + bi.z;
    float n3 = dw * rs * sc.w + bi.w;
    sx[tid * 4 + 0] = n0; sx[tid * 4 + 1] = n1;
    sx[tid * 4 + 2] = n2; sx[tid * 4 + 3] = n3;
    bf4 pk = { (__bf16)n0, (__bf16)n1, (__bf16)n2, (__bf16)n3 };
    *(bf4*)(xn + (size_t)t * Ddim + tid * 4) = pk;
    __syncthreads();

    // router: wave w computes logit for expert w
    const int wv = tid >> 5, lane = tid & 31;
    const float* rw = router_w + (size_t)wv * Ddim;
    float acc = 0.f;
    for (int j = lane; j < Ddim; j += 32) acc += sx[j] * rw[j];
    acc = wave_reduce_add(acc);
    if (lane == 0) slog[wv] = acc;
    __syncthreads();

    if (tid < Eexp) gate_out[(size_t)t * Eexp + tid] = slog[tid];

    if (tid == 0) {
        int   i1 = 0; float m1 = slog[0];
#pragma unroll
        for (int e = 1; e < Eexp; ++e) if (slog[e] > m1) { m1 = slog[e]; i1 = e; }
        int   i2 = -1; float m2 = -1e30f;
#pragma unroll
        for (int e = 0; e < Eexp; ++e) if (e != i1 && slog[e] > m2) { m2 = slog[e]; i2 = e; }
        const float mask = seq_mask[t];
        const float ez   = __expf(m2 - m1);
        const float inv  = 1.0f / (1.0f + ez);
        const float p1   = mask * inv;
        const float p2   = mask * ez * inv;

        int pos1 = atomicAdd(&counters[i1], 1);
        int g1   = atomicAdd(&counters[Eexp], 1);
        tok_list [i1 * Ntok + pos1] = t;
        slot_list[i1 * Ntok + pos1] = g1;
        prob_list[i1 * Ntok + pos1] = p1;

        int pos2 = atomicAdd(&counters[i2], 1);
        int g2   = atomicAdd(&counters[Eexp], 1);
        tok_list [i2 * Ntok + pos2] = t;
        slot_list[i2 * Ntok + pos2] = g2;
        prob_list[i2 * Ntok + pos2] = p2;
    }
}

// ---------------------------------------------------------------------------
// Kernel 2: per-expert tiled transpose + fp32 -> bf16 convert.
// src [z][R][C] fp32 -> dst [z][C][R] bf16. Block (32,8), 32x32 LDS tile.
// ---------------------------------------------------------------------------
__global__ __launch_bounds__(256) void transpose_bf16_kernel(
    const float* __restrict__ src, __bf16* __restrict__ dst, int R, int C)
{
    __shared__ float tile[32][33];
    const int z  = blockIdx.z;
    const int tx = threadIdx.x, ty = threadIdx.y;
    const int c0 = blockIdx.x * 32, r0 = blockIdx.y * 32;
    const float* s = src + (size_t)z * R * C;
    __bf16*      d = dst + (size_t)z * R * C;
#pragma unroll
    for (int i = 0; i < 4; ++i)
        tile[ty + i * 8][tx] = s[(size_t)(r0 + ty + i * 8) * C + c0 + tx];
    __syncthreads();
#pragma unroll
    for (int i = 0; i < 4; ++i)
        d[(size_t)(c0 + ty + i * 8) * R + r0 + tx] = (__bf16)tile[tx][ty + i * 8];
}

// ---------------------------------------------------------------------------
// Kernel 3: grouped GEMM1 + GELU. h[g] = gelu(xn[tok] @ w1[e] + b1[e]).
// grid (Hdim/256, Ntok/16, Eexp). 8 waves x (16x32 tile, 2 accumulators).
// ---------------------------------------------------------------------------
__global__ __launch_bounds__(256) void gemm1_kernel(
    const __bf16* __restrict__ xn, const __bf16* __restrict__ w1t,
    const float* __restrict__ b1, const int* __restrict__ counters,
    const int* __restrict__ tok_list, const int* __restrict__ slot_list,
    __bf16* __restrict__ hbuf)
{
    const int e   = blockIdx.z;
    const int cnt = counters[e];
    const int m0  = blockIdx.y * 16;
    if (m0 >= cnt) return;
    const int tid = threadIdx.x;

    __shared__ __bf16 sA[16][Ddim + 8];   // +8 bf16 pad -> conflict-free b128
    __shared__ int    sSlot[16];

    {   // stage 16 gathered token rows (16 x 1024 bf16)
        const int row   = tid >> 4;
        const int chunk = tid & 15;                  // 64 elements each
        const int p     = m0 + row;
        const int tok   = tok_list[e * Ntok + ((p < cnt) ? p : 0)];
        const uint4* src = (const uint4*)(xn + (size_t)tok * Ddim + chunk * 64);
        uint4*       dst = (uint4*)(&sA[row][chunk * 64]);
#pragma unroll
        for (int i = 0; i < 8; ++i) dst[i] = src[i];
        if (tid < 16)
            sSlot[tid] = (m0 + tid < cnt) ? slot_list[e * Ntok + m0 + tid] : -1;
    }
    __syncthreads();

    const int lane = tid & 31;
    const int wv   = tid >> 5;
    const int hi   = lane >> 4;        // K-half select
    const int lx   = lane & 15;        // M row (A) / N col (B,D)
    const int n0   = blockIdx.x * 256 + wv * 32;
    const __bf16* bbase = w1t + (size_t)e * Hdim * Ddim;
    const __bf16* bptr0 = bbase + (size_t)(n0 + lx) * Ddim;
    const __bf16* bptr1 = bbase + (size_t)(n0 + 16 + lx) * Ddim;

    v8f acc0 = { 0.f, 0.f, 0.f, 0.f, 0.f, 0.f, 0.f, 0.f };
    v8f acc1 = acc0;
#pragma unroll 2
    for (int k0 = 0; k0 < Ddim; k0 += 32) {
        ABFrag a, b0, b1f;
        a.u[0] = *(const uint4*)(&sA[lx][k0 + hi * 16]);
        a.u[1] = *(const uint4*)(&sA[lx][k0 + hi * 16 + 8]);
        const uint4* bp0 = (const uint4*)(bptr0 + k0 + hi * 16);
        const uint4* bp1 = (const uint4*)(bptr1 + k0 + hi * 16);
        b0.u[0] = bp0[0]; b0.u[1] = bp0[1];
        b1f.u[0] = bp1[0]; b1f.u[1] = bp1[1];
        acc0 = __builtin_amdgcn_wmma_f32_16x16x32_bf16(
            false, a.v, false, b0.v, (short)0, acc0, false, false);
        acc1 = __builtin_amdgcn_wmma_f32_16x16x32_bf16(
            false, a.v, false, b1f.v, (short)0, acc1, false, false);
    }

    const float bias0 = b1[e * Hdim + n0 + lx];
    const float bias1 = b1[e * Hdim + n0 + 16 + lx];
#pragma unroll
    for (int i = 0; i < 8; ++i) {
        const int M = hi ? (i + 8) : i;
        const int g = sSlot[M];
        if (g >= 0) {
            float v0 = gelu_exact(acc0[i] + bias0);
            float v1 = gelu_exact(acc1[i] + bias1);
            hbuf[(size_t)g * Hdim + n0 + lx]      = (__bf16)v0;
            hbuf[(size_t)g * Hdim + n0 + 16 + lx] = (__bf16)v1;
        }
    }
}

// ---------------------------------------------------------------------------
// Kernel 4: grouped GEMM2 + GELU + weighted combine into out (atomic f32 add).
// grid (Odim/256, Ntok/16, Eexp). 8 waves x (16x32 tile, 2 accumulators).
// ---------------------------------------------------------------------------
__global__ __launch_bounds__(256) void gemm2_kernel(
    const __bf16* __restrict__ hbuf, const __bf16* __restrict__ w2t,
    const float* __restrict__ b2, const int* __restrict__ counters,
    const int* __restrict__ tok_list, const int* __restrict__ slot_list,
    const float* __restrict__ prob_list, float* __restrict__ out)
{
    const int e   = blockIdx.z;
    const int cnt = counters[e];
    const int m0  = blockIdx.y * 16;
    if (m0 >= cnt) return;
    const int tid = threadIdx.x;

    __shared__ __bf16 sA[16][Hdim + 8];
    __shared__ int    sTok[16];
    __shared__ float  sProb[16];

    {   // stage 16 h-rows (16 x 2048 bf16)
        const int row   = tid >> 4;
        const int chunk = tid & 15;                  // 128 elements each
        const int p     = m0 + row;
        const int g     = slot_list[e * Ntok + ((p < cnt) ? p : 0)];
        const uint4* src = (const uint4*)(hbuf + (size_t)g * Hdim + chunk * 128);
        uint4*       dst = (uint4*)(&sA[row][chunk * 128]);
#pragma unroll
        for (int i = 0; i < 16; ++i) dst[i] = src[i];
        if (tid < 16) {
            const int  pp = m0 + tid;
            const bool ok = pp < cnt;
            sTok [tid] = ok ? tok_list [e * Ntok + pp] : -1;
            sProb[tid] = ok ? prob_list[e * Ntok + pp] : 0.f;
        }
    }
    __syncthreads();

    const int lane = tid & 31;
    const int wv   = tid >> 5;
    const int hi   = lane >> 4;
    const int lx   = lane & 15;
    const int n0   = blockIdx.x * 256 + wv * 32;
    const __bf16* bbase = w2t + (size_t)e * Odim * Hdim;
    const __bf16* bptr0 = bbase + (size_t)(n0 + lx) * Hdim;
    const __bf16* bptr1 = bbase + (size_t)(n0 + 16 + lx) * Hdim;

    v8f acc0 = { 0.f, 0.f, 0.f, 0.f, 0.f, 0.f, 0.f, 0.f };
    v8f acc1 = acc0;
#pragma unroll 2
    for (int k0 = 0; k0 < Hdim; k0 += 32) {
        ABFrag a, b0, b1f;
        a.u[0] = *(const uint4*)(&sA[lx][k0 + hi * 16]);
        a.u[1] = *(const uint4*)(&sA[lx][k0 + hi * 16 + 8]);
        const uint4* bp0 = (const uint4*)(bptr0 + k0 + hi * 16);
        const uint4* bp1 = (const uint4*)(bptr1 + k0 + hi * 16);
        b0.u[0] = bp0[0]; b0.u[1] = bp0[1];
        b1f.u[0] = bp1[0]; b1f.u[1] = bp1[1];
        acc0 = __builtin_amdgcn_wmma_f32_16x16x32_bf16(
            false, a.v, false, b0.v, (short)0, acc0, false, false);
        acc1 = __builtin_amdgcn_wmma_f32_16x16x32_bf16(
            false, a.v, false, b1f.v, (short)0, acc1, false, false);
    }

    const float bias0 = b2[e * Odim + n0 + lx];
    const float bias1 = b2[e * Odim + n0 + 16 + lx];
#pragma unroll
    for (int i = 0; i < 8; ++i) {
        const int M   = hi ? (i + 8) : i;
        const int tok = sTok[M];
        if (tok >= 0) {
            const float p = sProb[M];
            float y0 = gelu_exact(acc0[i] + bias0) * p;
            float y1 = gelu_exact(acc1[i] + bias1) * p;
            unsafeAtomicAdd(&out[(size_t)tok * Odim + n0 + lx],      y0);
            unsafeAtomicAdd(&out[(size_t)tok * Odim + n0 + 16 + lx], y1);
        }
    }
}

// ---------------------------------------------------------------------------
// Launch
// ---------------------------------------------------------------------------
extern "C" void kernel_launch(void* const* d_in, const int* in_sizes, int n_in,
                              void* d_out, int out_size, void* d_ws, size_t ws_size,
                              hipStream_t stream) {
    const float* x    = (const float*)d_in[0];
    const float* seq  = (const float*)d_in[1];
    const float* ln_s = (const float*)d_in[2];
    const float* ln_b = (const float*)d_in[3];
    const float* rw   = (const float*)d_in[4];
    const float* w1   = (const float*)d_in[5];
    const float* b1   = (const float*)d_in[6];
    const float* w2   = (const float*)d_in[7];
    const float* b2   = (const float*)d_in[8];

    float* out      = (float*)d_out;                       // [N, O]
    float* gate_out = out + (size_t)Ntok * Odim;           // [N, E]

    // workspace layout (bytes)
    char* ws = (char*)d_ws;
    size_t off = 0;
    __bf16* xn   = (__bf16*)(ws + off); off += (size_t)Ntok * Ddim * 2;        // 8 MB
    __bf16* w1t  = (__bf16*)(ws + off); off += (size_t)Eexp * Hdim * Ddim * 2; // 32 MB
    __bf16* w2t  = (__bf16*)(ws + off); off += (size_t)Eexp * Odim * Hdim * 2; // 32 MB
    __bf16* hbuf = (__bf16*)(ws + off); off += (size_t)(Ntok * 2) * Hdim * 2;  // 32 MB
    int*   tok_list  = (int*)(ws + off); off += (size_t)Eexp * Ntok * 4;
    int*   slot_list = (int*)(ws + off); off += (size_t)Eexp * Ntok * 4;
    float* prob_list = (float*)(ws + off); off += (size_t)Eexp * Ntok * 4;
    int*   counters  = (int*)(ws + off); off += (Eexp + 1) * 4;
    (void)ws_size; (void)in_sizes; (void)n_in; (void)out_size;

    hipMemsetAsync(out, 0, (size_t)Ntok * Odim * sizeof(float), stream);
    hipMemsetAsync(counters, 0, (Eexp + 1) * sizeof(int), stream);

    ln_router_kernel<<<Ntok, 256, 0, stream>>>(
        x, seq, ln_s, ln_b, rw, xn, gate_out, counters, tok_list, slot_list, prob_list);

    transpose_bf16_kernel<<<dim3(Hdim / 32, Ddim / 32, Eexp), dim3(32, 8), 0, stream>>>(
        w1, w1t, Ddim, Hdim);
    transpose_bf16_kernel<<<dim3(Odim / 32, Hdim / 32, Eexp), dim3(32, 8), 0, stream>>>(
        w2, w2t, Hdim, Odim);

    gemm1_kernel<<<dim3(Hdim / 256, Ntok / 16, Eexp), 256, 0, stream>>>(
        xn, w1t, b1, counters, tok_list, slot_list, hbuf);

    gemm2_kernel<<<dim3(Odim / 256, Ntok / 16, Eexp), 256, 0, stream>>>(
        hbuf, w2t, b2, counters, tok_list, slot_list, prob_list, out);
}
